// AtomicLinearTorch_16844861735163
// MI455X (gfx1250) — matrix operand order repaired and checked by reading, not compiled, not run on hardware
//
#include <hip/hip_runtime.h>

// y[b,o] = sum_k x[b,k]*w[o,k] + bias[o];  M=N=K=4096, fp32 in/out.
// Strategy: fp32 -> bf16 hi/lo split, 3x v_wmma_f32_16x16x32_bf16 per block,
// fp32 accumulate.  Fast path: pre-pass converts X/W into 4 bf16 planes in
// d_ws (128 MB, L2-resident), main GEMM stages tiles via TDM
// (tensor_load_to_lds) into double-buffered LDS, synced with s_wait_tensorcnt.
// Fallback (small ws): round-1 kernel with in-loop conversion.

#define NFEAT 4096
#define TILE_M 128
#define TILE_N 128
#define TILE_K 32
#define NPLANE ((size_t)NFEAT * (size_t)NFEAT)

typedef __attribute__((ext_vector_type(16))) __bf16 v16bf;
typedef __attribute__((ext_vector_type(8)))  __bf16 v8bf;
typedef __attribute__((ext_vector_type(8)))  float  v8f;
typedef __attribute__((ext_vector_type(4)))  float  v4f;
typedef __attribute__((ext_vector_type(4)))  unsigned int v4u;
typedef __attribute__((ext_vector_type(8)))  int    v8i;
typedef __attribute__((ext_vector_type(4)))  int    v4i;

// ---------------------------------------------------------------------------
// Shared WMMA micro-kernel: consume one 128(M)x32(K) X tile and 128(N)x32(K)
// W tile (bf16 hi/lo planes in LDS), update this wave's 4x2 C fragments.
// A frag: lane L<16 -> 16 contiguous K at a_k0.
// B frag: lane n<16 -> 16 contiguous K at b_k0.
// ---------------------------------------------------------------------------
__device__ __forceinline__ void compute_chunk(
    const __bf16* sXhi, const __bf16* sXlo,
    const __bf16* sWhi, const __bf16* sWlo,
    int wm, int wn, int lmod, int a_k0, int b_k0,
    v8f (&acc)[4][2])
{
    v16bf bhi[2], blo[2];
#pragma unroll
    for (int nf = 0; nf < 2; nf++) {
        int rowB = wn + nf * 16 + lmod;
        bhi[nf] = *(const v16bf*)(&sWhi[rowB * TILE_K + b_k0]);
        blo[nf] = *(const v16bf*)(&sWlo[rowB * TILE_K + b_k0]);
    }
#pragma unroll
    for (int mf = 0; mf < 4; mf++) {
        int rowA = wm + mf * 16 + lmod;
        v8bf ah0 = *(const v8bf*)(&sXhi[rowA * TILE_K + a_k0]);
        v8bf ah1 = *(const v8bf*)(&sXhi[rowA * TILE_K + a_k0 + 8]);
        v8bf al0 = *(const v8bf*)(&sXlo[rowA * TILE_K + a_k0]);
        v8bf al1 = *(const v8bf*)(&sXlo[rowA * TILE_K + a_k0 + 8]);
        v16bf ahi = __builtin_shufflevector(ah0, ah1,
            0,1,2,3,4,5,6,7,8,9,10,11,12,13,14,15);
        v16bf alo = __builtin_shufflevector(al0, al1,
            0,1,2,3,4,5,6,7,8,9,10,11,12,13,14,15);
#pragma unroll
        for (int nf = 0; nf < 2; nf++) {
            acc[mf][nf] = __builtin_amdgcn_wmma_f32_16x16x32_bf16(
                false, alo, false, bhi[nf], (short)0, acc[mf][nf], false, false);
            acc[mf][nf] = __builtin_amdgcn_wmma_f32_16x16x32_bf16(
                false, ahi, false, blo[nf], (short)0, acc[mf][nf], false, false);
            acc[mf][nf] = __builtin_amdgcn_wmma_f32_16x16x32_bf16(
                false, ahi, false, bhi[nf], (short)0, acc[mf][nf], false, false);
        }
    }
}

__device__ __forceinline__ void store_epilogue(
    const float* __restrict__ bias, float* __restrict__ out,
    int bm, int bn, int wm, int wn, int lmod, int l16,
    v8f (&acc)[4][2])
{
#pragma unroll
    for (int mf = 0; mf < 4; mf++) {
#pragma unroll
        for (int nf = 0; nf < 2; nf++) {
            int col  = bn + wn + nf * 16 + lmod;
            float bv = bias[col];
            int row0 = bm + wm + mf * 16 + l16 * 8;
#pragma unroll
            for (int r = 0; r < 8; r++) {
                __builtin_nontemporal_store(acc[mf][nf][r] + bv,
                    &out[(size_t)(row0 + r) * NFEAT + col]);
            }
        }
    }
}

// ---------------------------------------------------------------------------
// TDM tile load: 3D tensor (dim0=K elems, dim1=rows, dim2=planes), bf16.
// D# layout per CDNA5 ISA.
// ---------------------------------------------------------------------------
__device__ __forceinline__ void tdm_load_tile(
    unsigned lds_off, const void* gaddr,
    unsigned tile_d0, unsigned tile_d1, unsigned tile_d2,
    unsigned tensor_d0, unsigned tensor_d1, unsigned tensor_d2,
    unsigned long long d0_stride, unsigned long long d1_stride)
{
    unsigned long long ga = (unsigned long long)(uintptr_t)gaddr;
    v4u g0;
    g0[0] = 1u;                                        // count=1, user D#
    g0[1] = lds_off;                                   // lds_addr [63:32]
    g0[2] = (unsigned)(ga & 0xFFFFFFFFull);            // global_addr[31:0]
    g0[3] = (unsigned)((ga >> 32) & 0x1FFFFFFull)      // global_addr[56:32]
          | (2u << 30);                                // type=2 (image)
    v8i g1;
    g1[0] = (int)(1u << 16);                           // data_size=1 (2B), no mcast
    g1[1] = (int)(tensor_d0 << 16);                    // tensor_dim0[15:0]
    g1[2] = (int)((tensor_d0 >> 16) | (tensor_d1 << 16));
    g1[3] = (int)((tensor_d1 >> 16) | (tile_d0 << 16));
    g1[4] = (int)(tile_d1 | (tile_d2 << 16));          // tile_dim1, tile_dim2
    g1[5] = (int)(unsigned)(d0_stride & 0xFFFFFFFFull);
    g1[6] = (int)((unsigned)((d0_stride >> 32) & 0xFFFFull)
          | (unsigned)((d1_stride & 0xFFFFull) << 16));
    g1[7] = (int)(unsigned)((d1_stride >> 16) & 0xFFFFFFFFull);
    v4i g2;
    g2[0] = (int)tensor_d2;                            // tensor_dim2
    g2[1] = 0; g2[2] = 0; g2[3] = 0;                   // dim3 / tile_dim3 unused
    v4i g3 = {0, 0, 0, 0};
#if __clang_major__ >= 23
    v8i g1b = {0, 0, 0, 0, 0, 0, 0, 0};
    __builtin_amdgcn_tensor_load_to_lds(g0, g1, g2, g3, g1b, 0);
#else
    __builtin_amdgcn_tensor_load_to_lds(g0, g1, g2, g3, 0);
#endif
}

// ---------------------------------------------------------------------------
// Pre-pass: split fp32 X and W into bf16 hi/lo planes (ws layout:
// [Xhi][Xlo][Whi][Wlo], each NPLANE bf16).  Nontemporal fp32 reads (never
// needed again); bf16 planes stay hot in the 192MB L2.
// ---------------------------------------------------------------------------
__global__ __launch_bounds__(256) void split_fp32_to_bf16(
    const float* __restrict__ X, const float* __restrict__ W,
    __bf16* __restrict__ Xhi, __bf16* __restrict__ Xlo,
    __bf16* __restrict__ Whi, __bf16* __restrict__ Wlo)
{
    size_t i0 = ((size_t)blockIdx.x * 256 + threadIdx.x) * 8;
    v4f x0 = __builtin_nontemporal_load((const v4f*)(X + i0));
    v4f x1 = __builtin_nontemporal_load((const v4f*)(X + i0 + 4));
    v4f w0 = __builtin_nontemporal_load((const v4f*)(W + i0));
    v4f w1 = __builtin_nontemporal_load((const v4f*)(W + i0 + 4));
    v8bf xh, xl, wh, wl;
#pragma unroll
    for (int j = 0; j < 8; j++) {
        float xv = (j < 4) ? x0[j] : x1[j - 4];
        __bf16 hx = (__bf16)xv;
        xh[j] = hx;
        xl[j] = (__bf16)(xv - (float)hx);
        float wv = (j < 4) ? w0[j] : w1[j - 4];
        __bf16 hw = (__bf16)wv;
        wh[j] = hw;
        wl[j] = (__bf16)(wv - (float)hw);
    }
    *(v8bf*)(Xhi + i0) = xh;
    *(v8bf*)(Xlo + i0) = xl;
    *(v8bf*)(Whi + i0) = wh;
    *(v8bf*)(Wlo + i0) = wl;
}

// ---------------------------------------------------------------------------
// Fast path GEMM: bf16 planes in, TDM double-buffered LDS staging.
// sbuf[b] planes: 0=Xhi 1=Xlo 2=Whi 3=Wlo, each 128x32 bf16 (8KB).
// ---------------------------------------------------------------------------
__global__ __launch_bounds__(256) void linear_wmma_tdm(
    const __bf16* __restrict__ Xp,   // Xhi plane; Xlo at Xp+NPLANE
    const __bf16* __restrict__ Wp,   // Whi plane; Wlo at Wp+NPLANE
    const float* __restrict__ bias, float* __restrict__ out)
{
    __shared__ __bf16 sbuf[2][4][TILE_M * TILE_K];

    const int tid  = threadIdx.x;
    const int lane = tid & 31;
    const int wid  = tid >> 5;
    const int bm   = blockIdx.y * TILE_M;
    const int bn   = blockIdx.x * TILE_N;
    const int wm   = (wid >> 2) * 64;
    const int wn   = (wid & 3) * 32;
    const int l16  = lane >> 4;
    const int lmod = lane & 15;
    const int a_k0 = l16 ? 16 : 0;
    const int b_k0 = l16 ? 16 : 0;

    v8f acc[4][2];
#pragma unroll
    for (int i = 0; i < 4; i++)
#pragma unroll
        for (int j = 0; j < 2; j++)
#pragma unroll
            for (int e = 0; e < 8; e++) acc[i][j][e] = 0.0f;

    const bool issuer = (wid == 0);   // uniform per wave; wave0 drives the TDM
    auto issue = [&](int chunk, int b) {
        int kc = chunk * TILE_K;
        // X: 3D tensor {K=4096, rows=4096, planes=2(hi,lo)}, z-stride=NPLANE
        tdm_load_tile((unsigned)(size_t)&sbuf[b][0][0],
                      Xp + (size_t)bm * NFEAT + kc,
                      TILE_K, TILE_M, 2, NFEAT, NFEAT, 2,
                      (unsigned long long)NFEAT, (unsigned long long)NPLANE);
        // W: same shape
        tdm_load_tile((unsigned)(size_t)&sbuf[b][2][0],
                      Wp + (size_t)bn * NFEAT + kc,
                      TILE_K, TILE_N, 2, NFEAT, NFEAT, 2,
                      (unsigned long long)NFEAT, (unsigned long long)NPLANE);
    };

    const int NC = NFEAT / TILE_K;    // 128 chunks
    if (issuer) { issue(0, 0); issue(1, 1); }

    for (int i = 0; i < NC; ++i) {
        int b = i & 1;
        if (issuer) {
            if (i < NC - 1) __builtin_amdgcn_s_wait_tensorcnt((short)2);
            else            __builtin_amdgcn_s_wait_tensorcnt((short)0);
        }
        __syncthreads();              // chunk i resident in sbuf[b]
        compute_chunk(&sbuf[b][0][0], &sbuf[b][1][0],
                      &sbuf[b][2][0], &sbuf[b][3][0],
                      wm, wn, lmod, a_k0, b_k0, acc);
        __syncthreads();              // everyone done reading sbuf[b]
        if (issuer && i + 2 < NC) issue(i + 2, b);
    }

    store_epilogue(bias, out, bm, bn, wm, wn, lmod, l16, acc);
}

// ---------------------------------------------------------------------------
// Fallback GEMM (round-1 path): in-loop fp32->bf16 split, no ws.
// ---------------------------------------------------------------------------
__global__ __launch_bounds__(256) void linear_wmma_bf16x3(
    const float* __restrict__ X, const float* __restrict__ W,
    const float* __restrict__ bias, float* __restrict__ out)
{
    __shared__ __bf16 sXhi[TILE_M * TILE_K];
    __shared__ __bf16 sXlo[TILE_M * TILE_K];
    __shared__ __bf16 sWhi[TILE_N * TILE_K];
    __shared__ __bf16 sWlo[TILE_N * TILE_K];

    const int tid  = threadIdx.x;
    const int lane = tid & 31;
    const int wid  = tid >> 5;
    const int bm   = blockIdx.y * TILE_M;
    const int bn   = blockIdx.x * TILE_N;
    const int wm   = (wid >> 2) * 64;
    const int wn   = (wid & 3) * 32;
    const int l16  = lane >> 4;
    const int lmod = lane & 15;
    const int a_k0 = l16 ? 16 : 0;
    const int b_k0 = l16 ? 16 : 0;

    v8f acc[4][2];
#pragma unroll
    for (int i = 0; i < 4; i++)
#pragma unroll
        for (int j = 0; j < 2; j++)
#pragma unroll
            for (int e = 0; e < 8; e++) acc[i][j][e] = 0.0f;

    for (int kc = 0; kc < NFEAT; kc += TILE_K) {
#pragma unroll
        for (int i = 0; i < 4; i++) {
            int f   = tid + 256 * i;
            int row = f >> 3;
            int seg = f & 7;
            v4f xv = *(const v4f*)(X + (size_t)(bm + row) * NFEAT + kc + seg * 4);
            v4f wv = *(const v4f*)(W + (size_t)(bn + row) * NFEAT + kc + seg * 4);
            int base = row * TILE_K + seg * 4;
#pragma unroll
            for (int j = 0; j < 4; j++) {
                __bf16 hx = (__bf16)xv[j];
                sXhi[base + j] = hx;
                sXlo[base + j] = (__bf16)(xv[j] - (float)hx);
                __bf16 hw = (__bf16)wv[j];
                sWhi[base + j] = hw;
                sWlo[base + j] = (__bf16)(wv[j] - (float)hw);
            }
        }
        __syncthreads();
        compute_chunk(sXhi, sXlo, sWhi, sWlo, wm, wn, lmod, a_k0, b_k0, acc);
        __syncthreads();
    }

    store_epilogue(bias, out, bm, bn, wm, wn, lmod, l16, acc);
}

extern "C" void kernel_launch(void* const* d_in, const int* in_sizes, int n_in,
                              void* d_out, int out_size, void* d_ws, size_t ws_size,
                              hipStream_t stream) {
    const float* x    = (const float*)d_in[0];   // [4096,4096]
    const float* w    = (const float*)d_in[1];   // [4096,4096]
    const float* bias = (const float*)d_in[2];   // [4096]
    float* out        = (float*)d_out;           // [4096,4096]

    dim3 grid(NFEAT / TILE_N, NFEAT / TILE_M);   // 32 x 32
    dim3 block(256);

    const size_t need = 4 * NPLANE * sizeof(__bf16);   // 128 MB
    if (ws_size >= need) {
        __bf16* xhi = (__bf16*)d_ws;
        __bf16* xlo = xhi + NPLANE;
        __bf16* whi = xlo + NPLANE;
        __bf16* wlo = whi + NPLANE;
        split_fp32_to_bf16<<<(unsigned)(NPLANE / (256 * 8)), 256, 0, stream>>>(
            x, w, xhi, xlo, whi, wlo);
        linear_wmma_tdm<<<grid, block, 0, stream>>>(xhi, whi, bias, out);
    } else {
        linear_wmma_bf16x3<<<grid, block, 0, stream>>>(x, w, bias, out);
    }
}
